// Qwen3AttentionTP_57011395887274
// MI455X (gfx1250) — matrix operand order repaired and checked by reading, not compiled, not run on hardware
//
#include <hip/hip_runtime.h>
#include <hip/hip_bf16.h>
#include <stdint.h>

#define B_   2
#define S_   2048
#define H_   2048
#define NH_  16
#define D_   128
#define MTOT (B_*S_)   // 4096

typedef __attribute__((ext_vector_type(16))) __bf16 v16bf;
typedef __attribute__((ext_vector_type(8)))  float  v8f;
typedef unsigned short u16;
typedef int v4i __attribute__((vector_size(16)));   // 4 x i32 (b128 payload)

#ifndef __has_builtin
#define __has_builtin(x) 0
#endif
#if __has_builtin(__builtin_amdgcn_global_load_async_to_lds_b128)
#define HAS_ASYNC_LDS 1
#else
#define HAS_ASYNC_LDS 0
#endif
#if __has_builtin(__builtin_amdgcn_s_wait_asynccnt)
#define HAS_WAIT_ASYNC 1
#else
#define HAS_WAIT_ASYNC 0
#endif

typedef __attribute__((address_space(1))) v4i as1_v4i;   // global int4
typedef __attribute__((address_space(3))) v4i as3_v4i;   // LDS int4

// async global -> LDS 128-bit copy (per-lane addresses), ASYNCcnt tracked
__device__ __forceinline__ void async_ld_b128(const u16* g, u16* l) {
#if HAS_ASYNC_LDS
    __builtin_amdgcn_global_load_async_to_lds_b128((as1_v4i*)g, (as3_v4i*)l, 0, 0);
#else
    *(uint4*)l = *(const uint4*)g;
#endif
}
template <int N>
__device__ __forceinline__ void wait_async_lds() {
#if HAS_ASYNC_LDS
#if HAS_WAIT_ASYNC
    __builtin_amdgcn_s_wait_asynccnt((unsigned short)N);
#else
    asm volatile("s_wait_asynccnt %0" ::"i"(N) : "memory");
#endif
#endif
}

__device__ __forceinline__ u16 f32_bf16(float f) {
    unsigned int u = __float_as_uint(f);
    u += 0x7FFFu + ((u >> 16) & 1u);          // round-to-nearest-even
    return (u16)(u >> 16);
}
__device__ __forceinline__ float bf16_f32(u16 h) {
    return __uint_as_float(((unsigned int)h) << 16);
}

union FragU { v16bf v; uint4 q[2]; };

// A fragment (16x32, row-major source): lane holds row lane&15,
// K = {kh..kh+7, 16+kh..16+kh+7}, kh = (lane>>4)*8   (ISA 7.12.2)
__device__ __forceinline__ v16bf load_frag_a(const u16* base, int ld) {
    int lane = threadIdx.x & 31;
    int row  = lane & 15;
    int kh   = (lane >> 4) << 3;
    FragU f;
    f.q[0] = *(const uint4*)(base + row * ld + kh);
    f.q[1] = *(const uint4*)(base + row * ld + kh + 16);
    return f.v;
}
// B fragment (32x16): B[k][n] = Wrow[n][k]; lane holds column lane&15,
// K = {kb..kb+7, 16+kb..16+kb+7}, kb = (lane>>4)*8
__device__ __forceinline__ v16bf load_frag_b(const u16* base, int ld) {
    int lane = threadIdx.x & 31;
    int col  = lane & 15;
    int kb   = (lane >> 4) << 3;
    FragU f;
    f.q[0] = *(const uint4*)(base + col * ld + kb);
    f.q[1] = *(const uint4*)(base + col * ld + kb + 16);
    return f.v;
}
__device__ __forceinline__ v8f v8f_zero() {
    v8f z = {0.f,0.f,0.f,0.f,0.f,0.f,0.f,0.f};
    return z;
}
__device__ __forceinline__ v8f wmma_bf16(v16bf a, v16bf b, v8f c) {
    return __builtin_amdgcn_wmma_f32_16x16x32_bf16(false, a, false, b,
                                                   (short)0, c, false, false);
}

// ---------------- fp32 -> bf16 conversion ----------------
__global__ void cvt_bf16_kernel(const float* __restrict__ src,
                                u16* __restrict__ dst, int n) {
    int i = blockIdx.x * blockDim.x + threadIdx.x;
    int stride = gridDim.x * blockDim.x;
    for (; i < n; i += stride) dst[i] = f32_bf16(src[i]);
}

// ---------------- GEMM: C[M=4096,N=2048] = A @ W^T (+bias) ----------------
// Double-buffered LDS staging via async global->LDS copies.
// MODE 0: bf16 out, head-split [B,NH,S,D], bias added
// MODE 1: f32 out, row-major [M,N], no bias
template <int MODE>
__global__ __launch_bounds__(256) void gemm_kernel(const u16* __restrict__ A,
                                                   const u16* __restrict__ W,
                                                   const float* __restrict__ bias,
                                                   void* __restrict__ outp) {
    __shared__ u16 As[2][128][40];   // [buf][row m][k] (pad 40 -> 80B stride)
    __shared__ u16 Bs[2][128][40];   // [buf][row n][k]

    const int tid  = threadIdx.x;
    const int wave = tid >> 5;
    const int lane = tid & 31;
    const int ml = (wave >> 1) * 32;   // wave's 32-row strip inside tile
    const int nl = (wave & 1) * 64;    // wave's 64-col strip inside tile
    const int m0 = blockIdx.y * 128 + ml;
    const int n0 = blockIdx.x * 128 + nl;
    const u16* Ag = A + (size_t)blockIdx.y * 128 * H_;
    const u16* Wg = W + (size_t)blockIdx.x * 128 * H_;
    const int srow = tid >> 1;          // staging: row 0..127
    const int scol = (tid & 1) << 4;    // staging: k sub-offset 0/16

    v8f acc[2][4];
#pragma unroll
    for (int i = 0; i < 2; i++)
#pragma unroll
        for (int j = 0; j < 4; j++) acc[i][j] = v8f_zero();

    // prologue: stage k-tile 0 into buffer 0
    {
        const u16* a = Ag + (size_t)srow * H_ + scol;
        const u16* w = Wg + (size_t)srow * H_ + scol;
        async_ld_b128(a,     &As[0][srow][scol]);
        async_ld_b128(a + 8, &As[0][srow][scol + 8]);
        async_ld_b128(w,     &Bs[0][srow][scol]);
        async_ld_b128(w + 8, &Bs[0][srow][scol + 8]);
    }

    for (int k0 = 0; k0 < H_; k0 += 32) {
        const int buf = (k0 >> 5) & 1;
        if (k0 + 32 < H_) {            // stage next tile, then wait current
            const u16* a = Ag + (size_t)srow * H_ + (k0 + 32) + scol;
            const u16* w = Wg + (size_t)srow * H_ + (k0 + 32) + scol;
            async_ld_b128(a,     &As[buf ^ 1][srow][scol]);
            async_ld_b128(a + 8, &As[buf ^ 1][srow][scol + 8]);
            async_ld_b128(w,     &Bs[buf ^ 1][srow][scol]);
            async_ld_b128(w + 8, &Bs[buf ^ 1][srow][scol + 8]);
            wait_async_lds<4>();       // in-order: current tile resident
        } else {
            wait_async_lds<0>();
        }
        __syncthreads();

        v16bf af[2], bf[4];
#pragma unroll
        for (int mt = 0; mt < 2; mt++)
            af[mt] = load_frag_a(&As[buf][ml + mt * 16][0], 40);
#pragma unroll
        for (int nt = 0; nt < 4; nt++)
            bf[nt] = load_frag_b(&Bs[buf][nl + nt * 16][0], 40);
#pragma unroll
        for (int mt = 0; mt < 2; mt++)
#pragma unroll
            for (int nt = 0; nt < 4; nt++)
                acc[mt][nt] = wmma_bf16(af[mt], bf[nt], acc[mt][nt]);
        __syncthreads();               // safe to restage this buffer
    }

    const int rbase = (lane >> 4) << 3;
    const int cl    = lane & 15;
#pragma unroll
    for (int mt = 0; mt < 2; mt++)
#pragma unroll
        for (int nt = 0; nt < 4; nt++) {
            int n = n0 + nt * 16 + cl;
            float bv = (MODE == 0) ? bias[n] : 0.f;
#pragma unroll
            for (int r = 0; r < 8; r++) {
                int m = m0 + mt * 16 + rbase + r;
                float v = acc[mt][nt][r] + bv;
                if (MODE == 0) {
                    int bb = m >> 11, s = m & (S_ - 1);
                    int hd = n >> 7,  dd = n & (D_ - 1);
                    ((u16*)outp)[(((size_t)(bb * NH_ + hd) * S_ + s) << 7) + dd] =
                        f32_bf16(v);
                } else {
                    ((float*)outp)[(size_t)m * H_ + n] = v;
                }
            }
        }
}

// ---------------- RoPE (in-place, bf16, head-split layout) ----------------
__global__ void rope_kernel(u16* __restrict__ t, const float* __restrict__ cs,
                            const float* __restrict__ sn) {
    int idx = blockIdx.x * blockDim.x + threadIdx.x;  // B*NH*S*64 threads
    int j  = idx & 63;
    int s  = (idx >> 6) & (S_ - 1);
    int bh = idx >> 17;
    size_t base = (((size_t)bh * S_) + s) << 7;
    float x1 = bf16_f32(t[base + j]);
    float x2 = bf16_f32(t[base + j + 64]);
    float c1 = cs[s * D_ + j],      s1 = sn[s * D_ + j];
    float c2 = cs[s * D_ + j + 64], s2 = sn[s * D_ + j + 64];
    t[base + j]      = f32_bf16(x1 * c1 - x2 * s1);
    t[base + j + 64] = f32_bf16(x2 * c2 + x1 * s2);
}

// ---------------- Flash attention: per (head, 128-query block) ----------------
__global__ __launch_bounds__(256) void flash_kernel(const u16* __restrict__ Q,
                                                    const u16* __restrict__ K,
                                                    const u16* __restrict__ V,
                                                    u16* __restrict__ ctx) {
    __shared__ u16 Ks[128][136];     // K tile, row-major [n][d]
    __shared__ u16 VT[128][136];     // V tile transposed [d][n]
    __shared__ u16 Ps[8][16][136];   // per-wave P staging (C->A transpose)

    const int bh   = blockIdx.y;
    const int qb   = blockIdx.x;
    const int tid  = threadIdx.x;
    const int wave = tid >> 5, lane = tid & 31;
    const size_t hoff = (size_t)bh * S_ * D_;
    const int m0 = qb * 128 + wave * 16;
    const int rbase = (lane >> 4) << 3;
    const int cl    = lane & 15;

    v16bf qf[4];
#pragma unroll
    for (int kc = 0; kc < 4; kc++)
        qf[kc] = load_frag_a(Q + hoff + (size_t)m0 * D_ + kc * 32, D_);

    float mrow[8], lrow[8];
    v8f oacc[8];
#pragma unroll
    for (int r = 0; r < 8; r++) { mrow[r] = -INFINITY; lrow[r] = 0.f; }
#pragma unroll
    for (int dt = 0; dt < 8; dt++) oacc[dt] = v8f_zero();

    const float scale = 0.08838834764831845f;  // 1/sqrt(128)

    for (int jb = 0; jb < S_ / 128; ++jb) {
        // ---- stage K (async, row-major) and V (transposed scatter) ----
        {
            const u16* kg = K + hoff + (size_t)jb * 128 * D_;
            const u16* vg = V + hoff + (size_t)jb * 128 * D_;
            int c = (tid & 15) * 8;
#pragma unroll
            for (int i = 0; i < 8; i++) {
                int r = (tid >> 4) + i * 16;
                async_ld_b128(kg + r * D_ + c, &Ks[r][c]);
                uint4 vv = *(const uint4*)(vg + r * D_ + c);
                const u16* pv = (const u16*)&vv;
#pragma unroll
                for (int j = 0; j < 8; j++) VT[c + j][r] = pv[j];
            }
        }
        wait_async_lds<0>();
        __syncthreads();

        // ---- S = Q @ K^T ----
        v8f sacc[8];
#pragma unroll
        for (int nt = 0; nt < 8; nt++) sacc[nt] = v8f_zero();
#pragma unroll
        for (int nt = 0; nt < 8; nt++)
#pragma unroll
            for (int kc = 0; kc < 4; kc++) {
                v16bf bf = load_frag_b(&Ks[nt * 16][0] + kc * 32, 136);
                sacc[nt] = wmma_bf16(qf[kc], bf, sacc[nt]);
            }

        // ---- online softmax (each 16-lane half owns 8 rows) ----
#pragma unroll
        for (int r = 0; r < 8; r++) {
            float t = sacc[0][r] * scale;
#pragma unroll
            for (int nt = 1; nt < 8; nt++) t = fmaxf(t, sacc[nt][r] * scale);
#pragma unroll
            for (int off = 1; off < 16; off <<= 1)
                t = fmaxf(t, __shfl_xor(t, off, 32));
            float mnew  = fmaxf(mrow[r], t);
            float alpha = __expf(mrow[r] - mnew);
            mrow[r] = mnew;
            float sum = 0.f;
#pragma unroll
            for (int nt = 0; nt < 8; nt++) {
                float p = __expf(sacc[nt][r] * scale - mnew);
                sacc[nt][r] = p;
                sum += p;
            }
#pragma unroll
            for (int off = 1; off < 16; off <<= 1)
                sum += __shfl_xor(sum, off, 32);
            lrow[r] = lrow[r] * alpha + sum;
#pragma unroll
            for (int dt = 0; dt < 8; dt++) oacc[dt][r] *= alpha;
        }

        // ---- transpose P via per-wave LDS slice (C layout -> A layout) ----
#pragma unroll
        for (int nt = 0; nt < 8; nt++)
#pragma unroll
            for (int r = 0; r < 8; r++)
                Ps[wave][rbase + r][nt * 16 + cl] = f32_bf16(sacc[nt][r]);
        asm volatile("s_wait_dscnt 0" ::: "memory");

        v16bf pf[4];
#pragma unroll
        for (int kc = 0; kc < 4; kc++)
            pf[kc] = load_frag_a(&Ps[wave][0][0] + kc * 32, 136);

        // ---- O += P @ V ----
#pragma unroll
        for (int dt = 0; dt < 8; dt++)
#pragma unroll
            for (int kc = 0; kc < 4; kc++) {
                v16bf bf = load_frag_b(&VT[dt * 16][0] + kc * 32, 136);
                oacc[dt] = wmma_bf16(pf[kc], bf, oacc[dt]);
            }
        __syncthreads();
    }

    // ---- normalize + write ctx (merge heads: [B*S, H] bf16) ----
    int bb = bh >> 4, hd = bh & 15;
#pragma unroll
    for (int dt = 0; dt < 8; dt++)
#pragma unroll
        for (int r = 0; r < 8; r++) {
            int srow = m0 + rbase + r;
            float v = oacc[dt][r] / lrow[r];
            ctx[((size_t)(bb * S_ + srow) * H_) + hd * D_ + dt * 16 + cl] =
                f32_bf16(v);
        }
}

extern "C" void kernel_launch(void* const* d_in, const int* in_sizes, int n_in,
                              void* d_out, int out_size, void* d_ws, size_t ws_size,
                              hipStream_t stream) {
    const float* x  = (const float*)d_in[0];
    const float* cs = (const float*)d_in[1];
    const float* sn = (const float*)d_in[2];
    const float* wq = (const float*)d_in[3];
    const float* bq = (const float*)d_in[4];
    const float* wk = (const float*)d_in[5];
    const float* bk = (const float*)d_in[6];
    const float* wv = (const float*)d_in[7];
    const float* bv = (const float*)d_in[8];
    const float* wo = (const float*)d_in[9];
    float* out = (float*)d_out;

    char* ws = (char*)d_ws;
    u16* xb  = (u16*)(ws);                      // 16 MB  [4096,2048] bf16
    u16* wqb = (u16*)(ws + (size_t)(16 << 20)); // 8 MB each
    u16* wkb = (u16*)(ws + (size_t)(24 << 20));
    u16* wvb = (u16*)(ws + (size_t)(32 << 20));
    u16* wob = (u16*)(ws + (size_t)(40 << 20));
    u16* qh  = (u16*)(ws + (size_t)(48 << 20)); // 16 MB each [B,NH,S,D]
    u16* kh  = (u16*)(ws + (size_t)(64 << 20));
    u16* vh  = (u16*)(ws + (size_t)(80 << 20));
    u16* ctx = (u16*)(ws + (size_t)(96 << 20)); // 16 MB  [4096,2048]

    cvt_bf16_kernel<<<4096, 256, 0, stream>>>(x,  xb,  MTOT * H_);
    cvt_bf16_kernel<<<4096, 256, 0, stream>>>(wq, wqb, H_ * H_);
    cvt_bf16_kernel<<<4096, 256, 0, stream>>>(wk, wkb, H_ * H_);
    cvt_bf16_kernel<<<4096, 256, 0, stream>>>(wv, wvb, H_ * H_);
    cvt_bf16_kernel<<<4096, 256, 0, stream>>>(wo, wob, H_ * H_);

    dim3 g(H_ / 128, MTOT / 128), blk(256);
    gemm_kernel<0><<<g, blk, 0, stream>>>(xb, wqb, bq, qh);
    gemm_kernel<0><<<g, blk, 0, stream>>>(xb, wkb, bk, kh);
    gemm_kernel<0><<<g, blk, 0, stream>>>(xb, wvb, bv, vh);

    int rope_threads = B_ * NH_ * S_ * 64;
    rope_kernel<<<rope_threads / 256, 256, 0, stream>>>(qh, cs, sn);
    rope_kernel<<<rope_threads / 256, 256, 0, stream>>>(kh, cs, sn);

    flash_kernel<<<dim3(S_ / 128, B_ * NH_), blk, 0, stream>>>(qh, kh, vh, ctx);

    gemm_kernel<1><<<g, blk, 0, stream>>>(ctx, wob, nullptr, out);
}
